// LSTM_66219805770265
// MI455X (gfx1250) — 16-bit and 8-bit matrix operand order repaired and checked by reading, compiled, not run on hardware
//
#include <hip/hip_runtime.h>
#include <hip/hip_bf16.h>


#define T_ 1024
#define B_ 64
#define I_ 256
#define H_ 256
#define L_ 2

#define USE_HW_CVT  1
#define USE_HW_TANH 1

typedef __attribute__((ext_vector_type(16))) __bf16 v16bf;
typedef __attribute__((ext_vector_type(8)))  float  v8f;
typedef __attribute__((ext_vector_type(8)))  int    v8i;

__device__ inline unsigned short f32_to_bf16_sw(float f) {
  unsigned u = __float_as_uint(f);
  u += 0x7fffu + ((u >> 16) & 1u);
  return (unsigned short)(u >> 16);
}
__device__ inline unsigned char f32_to_e4m3_sw(float f) {
  unsigned sgn = (__float_as_uint(f) >> 24) & 0x80u;
  float a = fabsf(f);
  if (!(a == a)) return (unsigned char)(sgn | 0x7Fu);
  if (a >= 448.f) return (unsigned char)(sgn | 0x7Eu);
  if (a < 0.015625f) {
    int M = (int)rintf(a * 512.f);
    if (M > 7) return (unsigned char)(sgn | 0x08u);
    return (unsigned char)(sgn | (unsigned)M);
  }
  int e; float m = frexpf(a, &e);
  int E = e - 1;
  int M = (int)rintf(m * 16.f) - 8;
  if (M == 8) { M = 0; ++E; }
  if (E > 8) return (unsigned char)(sgn | 0x7Eu);
  return (unsigned char)(sgn | (unsigned)((E + 7) << 3) | (unsigned)M);
}

__device__ inline unsigned cvt_pk_bf16(float a, float b) {
#if USE_HW_CVT
  unsigned d;
  asm volatile("v_cvt_pk_bf16_f32 %0, %1, %2" : "=v"(d) : "v"(a), "v"(b));
  return d;
#else
  return (unsigned)f32_to_bf16_sw(a) | ((unsigned)f32_to_bf16_sw(b) << 16);
#endif
}
__device__ inline unsigned cvt_pk_fp8(float a, float b) {
#if USE_HW_CVT
  unsigned d;
  asm volatile("v_cvt_pk_fp8_f32 %0, %1, %2" : "=v"(d) : "v"(a), "v"(b));
  return d;
#else
  return (unsigned)f32_to_e4m3_sw(a) | ((unsigned)f32_to_e4m3_sw(b) << 8);
#endif
}
__device__ inline float tanh_hw(float x) {
#if USE_HW_TANH
  float r;
  asm volatile("v_tanh_f32 %0, %1\n\tv_nop" : "=v"(r) : "v"(x));
  return r;
#else
  float cx = fminf(fmaxf(x, -12.f), 12.f);
  float e = __expf(2.f * cx);
  return 1.f - 2.f / (e + 1.f);
#endif
}

__device__ inline float bf_lo(int u) { return __uint_as_float(((unsigned)u) << 16); }
__device__ inline float bf_hi(int u) { return __uint_as_float(((unsigned)u) & 0xffff0000u); }

__global__ __launch_bounds__(256) void lstm_xproj(
    const float* __restrict__ x, const float* __restrict__ W,
    const float* __restrict__ bias, unsigned short* __restrict__ xpF) {
  __shared__ unsigned short As[64 * 32];
  __shared__ unsigned short Bs[256 * 32];

  const int t    = blockIdx.x;
  const int nb   = blockIdx.y;
  const int g    = nb >> 1, l = nb & 1;
  const int tid  = threadIdx.x;
  const int lane = tid & 31;
  const int w    = tid >> 5;
  const int mt   = w & 3;
  const int ng   = w >> 2;

  v8f acc[8];
  const float* brow = bias + (g * L_ + l) * H_;
  #pragma unroll
  for (int j = 0; j < 8; ++j) {
    float bv = brow[ng * 128 + j * 16 + (lane & 15)];
    #pragma unroll
    for (int r = 0; r < 8; ++r) acc[j][r] = bv;
  }

  const float* xA = x + (size_t)(t * B_) * I_;
  const float* wB = W + (size_t)((g * L_ + l) * H_) * I_;

  for (int kc = 0; kc < 8; ++kc) {
    const int k0 = kc * 32;
    __syncthreads();
    {
      int row = tid >> 2, col = (tid & 3) * 8;
      const float4* s4 = (const float4*)(xA + (size_t)row * I_ + k0 + col);
      float4 a0 = s4[0], a1 = s4[1];
      unsigned* dst = (unsigned*)(As + row * 32 + col);
      dst[0] = cvt_pk_bf16(a0.x, a0.y);
      dst[1] = cvt_pk_bf16(a0.z, a0.w);
      dst[2] = cvt_pk_bf16(a1.x, a1.y);
      dst[3] = cvt_pk_bf16(a1.z, a1.w);
    }
    {
      const float4* s4 = (const float4*)(wB + (size_t)tid * I_ + k0);
      unsigned* dst = (unsigned*)(Bs + tid * 32);
      #pragma unroll
      for (int q = 0; q < 8; q += 2) {
        float4 b0 = s4[q], b1 = s4[q + 1];
        dst[2 * q + 0] = cvt_pk_bf16(b0.x, b0.y);
        dst[2 * q + 1] = cvt_pk_bf16(b0.z, b0.w);
        dst[2 * q + 2] = cvt_pk_bf16(b1.x, b1.y);
        dst[2 * q + 3] = cvt_pk_bf16(b1.z, b1.w);
      }
    }
    __syncthreads();

    union { v16bf v; int4 i[2]; } fa;
    {
      int row = mt * 16 + (lane & 15);
      int kl  = (lane >> 4) * 8;
      fa.i[0] = *(const int4*)(As + row * 32 + kl);
      fa.i[1] = *(const int4*)(As + row * 32 + kl + 16);
    }
    #pragma unroll
    for (int j = 0; j < 8; ++j) {
      union { v16bf v; int4 i[2]; } fb;
      int n  = ng * 128 + j * 16 + (lane & 15);
      int kb = (lane >> 4) * 8;
      fb.i[0] = *(const int4*)(Bs + n * 32 + kb);
      fb.i[1] = *(const int4*)(Bs + n * 32 + kb + 16);
      acc[j] = __builtin_amdgcn_wmma_f32_16x16x32_bf16(
          false, fa.v, false, fb.v, (short)0, acc[j], false, false);
    }
  }

  #pragma unroll
  for (int j = 0; j < 8; ++j) {
    int jt = ng * 8 + j;
    size_t off = (((((size_t)(t * L_ + l) * 4 + g) * 4 + mt) * 16) + jt) * 256
                 + (size_t)lane * 8;
    int4 pk;
    pk.x = (int)cvt_pk_bf16(16.f * acc[j][0], 16.f * acc[j][1]);
    pk.y = (int)cvt_pk_bf16(16.f * acc[j][2], 16.f * acc[j][3]);
    pk.z = (int)cvt_pk_bf16(16.f * acc[j][4], 16.f * acc[j][5]);
    pk.w = (int)cvt_pk_bf16(16.f * acc[j][6], 16.f * acc[j][7]);
    *(int4*)(xpF + off) = pk;
  }
}

__global__ __launch_bounds__(512) void lstm_recur(
    const unsigned short* __restrict__ xpF, const float* __restrict__ h0,
    const float* __restrict__ c0, const float* __restrict__ U,
    float* __restrict__ out) {
  extern __shared__ unsigned char smem[];
  unsigned char* U8  = smem;
  unsigned char* hF0 = smem + 4 * H_ * H_;
  unsigned char* hF1 = hF0 + 16 * H_;

  const int blk  = blockIdx.x;
  const int l    = blk >> 2;
  const int bblk = blk & 3;
  const int b0   = bblk * 16;
  const int tid  = threadIdx.x;
  const int lane = tid & 31;
  const int w    = tid >> 5;
  const int j0   = w * 16;
  const int col  = lane & 15;
  const int rhi  = (lane >> 4) * 8;

  for (int e = tid * 2; e < 4 * H_ * H_; e += 1024) {
    int g = e >> 16, n = (e >> 8) & 255, k = e & 255;
    const float* up = U + (((size_t)g * L_ + l) * H_ + n) * H_ + k;
    unsigned pk = cvt_pk_fp8(up[0] * 16.f, up[1] * 16.f);
    *(unsigned short*)(U8 + e) = (unsigned short)pk;
  }
  {
    int e = tid * 8, mm = e >> 8, kk0 = e & 255;
    const float* hp = h0 + ((size_t)l * B_ + b0 + mm) * H_ + kk0;
    unsigned q0 = cvt_pk_fp8(hp[0], hp[1]);
    unsigned q1 = cvt_pk_fp8(hp[2], hp[3]);
    unsigned q2 = cvt_pk_fp8(hp[4], hp[5]);
    unsigned q3 = cvt_pk_fp8(hp[6], hp[7]);
    int cch = kk0 >> 6, rem = kk0 & 63;
    int p = rem >> 4, hi = (rem >> 3) & 1;
    unsigned char* dst = hF0 + cch * 1024 + (mm + 16 * hi) * 32 + p * 8;
    *(unsigned*)(dst)     = (q0 & 0xffffu) | (q1 << 16);
    *(unsigned*)(dst + 4) = (q2 & 0xffffu) | (q3 << 16);
  }
  float c[8];
  #pragma unroll
  for (int r = 0; r < 8; ++r)
    c[r] = c0[((size_t)l * B_ + b0 + rhi + r) * H_ + j0 + col];
  float hv[8];
  __syncthreads();

  const int n    = j0 + col;
  const int wrem = n & 63;
  const int wbase = (n >> 6) * 1024 + ((wrem >> 3) & 1) * 512 +
                    (wrem >> 4) * 8 + (n & 7);
  const float s  = 1.f / 16.f;
  const float sh = 1.f / 32.f;
  const size_t tlStride = (size_t)L_ * 4 * 4 * 4096;

  #pragma unroll 1
  for (int t = 0; t < T_; ++t) {
    const unsigned char* hR = (t & 1) ? hF1 : hF0;
    unsigned char*       hW = (t & 1) ? hF0 : hF1;

    v8i A[4];
    #pragma unroll
    for (int cch = 0; cch < 4; ++cch)
      A[cch] = *(const v8i*)(hR + cch * 1024 + lane * 32);

    const size_t tlBase = ((size_t)t * L_ + l) * (4 * 4 * 4096);
    v8f acc[4];
    #pragma unroll
    for (int g = 0; g < 4; ++g) {
      const unsigned short* pg =
          xpF + tlBase + (size_t)g * 16384 + (size_t)bblk * 4096 + w * 256 + lane * 8;
      int4 d = *(const int4*)pg;
      __builtin_prefetch(pg + tlStride, 0, 0);
      acc[g][0] = bf_lo(d.x); acc[g][1] = bf_hi(d.x);
      acc[g][2] = bf_lo(d.y); acc[g][3] = bf_hi(d.y);
      acc[g][4] = bf_lo(d.z); acc[g][5] = bf_hi(d.z);
      acc[g][6] = bf_lo(d.w); acc[g][7] = bf_hi(d.w);
    }

    #pragma unroll
    for (int cch = 0; cch < 4; ++cch) {
      #pragma unroll
      for (int g = 0; g < 4; ++g) {
        union { v8i v; int2 d[4]; } Bf;
        const unsigned char* bb =
            U8 + ((size_t)g * H_ + j0 + col) * H_ + cch * 64 + (lane >> 4) * 8;
        Bf.d[0] = *(const int2*)(bb);
        Bf.d[1] = *(const int2*)(bb + 16);
        Bf.d[2] = *(const int2*)(bb + 32);
        Bf.d[3] = *(const int2*)(bb + 48);
        acc[g] = __builtin_amdgcn_wmma_f32_16x16x64_fp8_fp8(
            A[cch], Bf.v, (short)0, acc[g], false, false);
      }
    }

    #pragma unroll
    for (int r = 0; r < 8; ++r) {
      float gf = fmaf(0.5f, tanh_hw(acc[0][r] * sh), 0.5f);
      float gi = fmaf(0.5f, tanh_hw(acc[1][r] * sh), 0.5f);
      float go = fmaf(0.5f, tanh_hw(acc[2][r] * sh), 0.5f);
      float gc = tanh_hw(acc[3][r] * s);
      float cn = gf * c[r] + gi * gc;
      c[r]  = cn;
      hv[r] = go * tanh_hw(cn);
    }

    {
      unsigned p01 = cvt_pk_fp8(hv[0], hv[1]);
      unsigned p23 = cvt_pk_fp8(hv[2], hv[3]);
      unsigned p45 = cvt_pk_fp8(hv[4], hv[5]);
      unsigned p67 = cvt_pk_fp8(hv[6], hv[7]);
      unsigned char* hw0 = hW + wbase + rhi * 32;
      hw0[0 * 32] = (unsigned char)(p01 & 0xff);
      hw0[1 * 32] = (unsigned char)((p01 >> 8) & 0xff);
      hw0[2 * 32] = (unsigned char)(p23 & 0xff);
      hw0[3 * 32] = (unsigned char)((p23 >> 8) & 0xff);
      hw0[4 * 32] = (unsigned char)(p45 & 0xff);
      hw0[5 * 32] = (unsigned char)((p45 >> 8) & 0xff);
      hw0[6 * 32] = (unsigned char)(p67 & 0xff);
      hw0[7 * 32] = (unsigned char)((p67 >> 8) & 0xff);
    }
    if (l == 1) {
      float* o = out + ((size_t)t * B_ + b0 + rhi) * H_ + j0 + col;
      #pragma unroll
      for (int r = 0; r < 8; ++r) o[(size_t)r * H_] = hv[r];
    }
    __syncthreads();
  }

  float* hn_out = out + (size_t)T_ * B_ * H_ +
                  ((size_t)l * B_ + b0 + rhi) * H_ + j0 + col;
  float* cn_out = hn_out + (size_t)L_ * B_ * H_;
  #pragma unroll
  for (int r = 0; r < 8; ++r) {
    hn_out[(size_t)r * H_] = hv[r];
    cn_out[(size_t)r * H_] = c[r];
  }
}

extern "C" void kernel_launch(void* const* d_in, const int* in_sizes, int n_in,
                              void* d_out, int out_size, void* d_ws, size_t ws_size,
                              hipStream_t stream) {
  (void)in_sizes; (void)n_in; (void)out_size; (void)ws_size;
  const float* x  = (const float*)d_in[0];
  const float* h0 = (const float*)d_in[1];
  const float* c0 = (const float*)d_in[2];
  const float* W  = (const float*)d_in[3];
  const float* U  = (const float*)d_in[4];
  const float* b  = (const float*)d_in[5];
  float* out = (float*)d_out;
  unsigned short* xpF = (unsigned short*)d_ws;

  dim3 g1(T_, 8);
  lstm_xproj<<<g1, 256, 0, stream>>>(x, W, b, xpF);

  const int smem = 4 * H_ * H_ + 2 * 16 * H_;
  (void)hipFuncSetAttribute((const void*)lstm_recur,
                            hipFuncAttributeMaxDynamicSharedMemorySize, smem);
  lstm_recur<<<8, 512, smem, stream>>>(xpF, h0, c0, U, out);
}
